// BiPhaseScorer_13924283973715
// MI455X (gfx1250) — matrix operand order repaired and checked by reading, not compiled, not run on hardware
//
#include <hip/hip_runtime.h>
#include <hip/hip_bf16.h>
#include <math.h>
#include <stdint.h>

typedef __bf16 bf16;
typedef __attribute__((ext_vector_type(16))) __bf16 v16bf;
typedef __attribute__((ext_vector_type(8)))  __bf16 v8bf;
typedef __attribute__((ext_vector_type(8)))  float  v8f;
typedef __attribute__((ext_vector_type(4)))  unsigned int u32x4;
typedef __attribute__((ext_vector_type(8)))  int i32x8;
typedef __attribute__((ext_vector_type(4)))  int i32x4;

#define BDIM 2
#define SDIM 2048
#define EDIM 512
#define HDIM 8
#define DDIM 64
#define FDIM 192   // 3*D polar feature dim

#if __has_builtin(__builtin_amdgcn_tensor_load_to_lds)
#define USE_TDM 1
#else
#define USE_TDM 0
#endif

// ---------------- WMMA helpers (CDNA5 bf16 16x16x32, f32 accum) ----------------

static __device__ __forceinline__ v8f wmma_bf(v16bf a, v16bf b, v8f c) {
  // 8 args: (neg_a, A, neg_b, B, c_mod, C, reuse_a, reuse_b)
  return __builtin_amdgcn_wmma_f32_16x16x32_bf16(false, a, false, b, (short)0, c,
                                                 false, false);
}

// A-matrix bf16 16x32: lane holds row m=lane%16; k runs: h*8..h*8+7 and 16+h*8..16+h*8+7
static __device__ __forceinline__ v16bf ldA_bf16(const bf16* base, int ldm, int lane) {
  int m = lane & 15, h = (lane >> 4) & 1;
  v8bf lo = *(const v8bf*)(base + (size_t)m * ldm + h * 8);
  v8bf hi = *(const v8bf*)(base + (size_t)m * ldm + 16 + h * 8);
  v16bf a;
#pragma unroll
  for (int i = 0; i < 8; ++i) { a[i] = lo[i]; a[i + 8] = hi[i]; }
  return a;
}

static __device__ __forceinline__ v16bf ldA_f32(const float* base, int ldm, int lane) {
  int m = lane & 15, h = (lane >> 4) & 1;
  const float* p0 = base + (size_t)m * ldm + h * 8;
  const float* p1 = base + (size_t)m * ldm + 16 + h * 8;
  v16bf a;
#pragma unroll
  for (int i = 0; i < 8; ++i) { a[i] = (__bf16)p0[i]; a[i + 8] = (__bf16)p1[i]; }
  return a;
}

// B-matrix bf16 32x16: lane holds col n=lane%16; k runs: h*8..h*8+7 and 16+h*8..16+h*8+7.
// "base" points at element (n=0, k=0) of memory laid out value(n,k) = base[n*ldn + k].
static __device__ __forceinline__ v16bf ldB_bf16(const bf16* base, int ldn, int lane) {
  int n = lane & 15, kh = (lane >> 4) & 1;
  const bf16* p = base + (size_t)n * ldn + kh * 8;
  v8bf b0 = *(const v8bf*)p, b1 = *(const v8bf*)(p + 16);
  v16bf b;
#pragma unroll
  for (int i = 0; i < 8; ++i) { b[i] = b0[i]; b[i + 8] = b1[i]; }
  return b;
}

static __device__ __forceinline__ v16bf ldB_f32(const float* base, int ldn, int lane) {
  int n = lane & 15, kh = (lane >> 4) & 1;
  const float* p = base + (size_t)n * ldn + kh * 8;
  v16bf b;
#pragma unroll
  for (int i = 0; i < 16; ++i) b[i] = (__bf16)p[i < 8 ? i : i + 8];
  return b;
}

#if USE_TDM
// ---------------- Tensor Data Mover: 2D tile (global -> LDS), ISA D# layout ----------------
// data_size_log: 0=1B,1=2B,2=4B,3=8B. Dims/strides in data_size units.
static __device__ __forceinline__ void tdm_load_2d(
    unsigned lds_off, const void* gptr,
    unsigned tensor_d0, unsigned tensor_d1,
    unsigned tile_d0, unsigned tile_d1,
    unsigned stride_d0, unsigned data_size_log) {
  unsigned long long ga = (unsigned long long)(uintptr_t)gptr;
  u32x4 g0;
  g0[0] = 1u;                                            // count=1, user descriptor
  g0[1] = lds_off;                                       // lds_addr [63:32]
  g0[2] = (unsigned)(ga & 0xFFFFFFFFu);                  // global_addr [95:64]
  g0[3] = (unsigned)((ga >> 32) & 0x01FFFFFFu) | (2u << 30); // addr[56:32] | type=2
  i32x8 g1;
  g1[0] = (int)(data_size_log << 16);                    // wg_mask=0, data_size
  g1[1] = (int)((tensor_d0 & 0xFFFFu) << 16);            // tensor_dim0[15:0] at [63:48]
  g1[2] = (int)(((tensor_d0 >> 16) & 0xFFFFu) | ((tensor_d1 & 0xFFFFu) << 16));
  g1[3] = (int)(((tensor_d1 >> 16) & 0xFFFFu) | ((tile_d0 & 0xFFFFu) << 16));
  g1[4] = (int)(tile_d1 & 0xFFFFu);                      // tile_dim1; tile_dim2=0 (2D)
  g1[5] = (int)stride_d0;                                // tensor_dim0_stride[31:0]
  g1[6] = 0;                                             // stride[47:32]=0; dim1_stride lo=0
  g1[7] = 0;
  i32x4 gz = {0, 0, 0, 0};
#if __clang_major__ >= 23
  i32x8 gz8 = {0, 0, 0, 0, 0, 0, 0, 0};
  __builtin_amdgcn_tensor_load_to_lds(g0, g1, gz, gz, gz8, 0);
#else
  __builtin_amdgcn_tensor_load_to_lds(g0, g1, gz, gz, 0);
#endif
}
#endif

// ---------------- Kernel 1: dual projection + polar featurization ----------------
// X1@W1^T+b1 -> x, X2@W2^T+b2 -> y ; r=sqrt(x^2+y^2); feat=[cs*x/r, cs*y/r, rs*r]
// feat layout: [B,H,S,FDIM] bf16. Block: 128 thr (4 waves), tile 64 rows x 64 cols (one head).
__global__ __launch_bounds__(128) void k_proj_polar(
    const float* __restrict__ X1, const float* __restrict__ X2,
    const float* __restrict__ W1, const float* __restrict__ b1,
    const float* __restrict__ W2, const float* __restrict__ b2,
    bf16* __restrict__ feat, float cs, float rs) {
  int lane = threadIdx.x & 31, wave = threadIdx.x >> 5;
  int m0 = blockIdx.x * 64 + wave * 16;
  int h  = blockIdx.y;
  v8f zero = {};
  v8f accx[4], accy[4];
#pragma unroll
  for (int n = 0; n < 4; ++n) { accx[n] = zero; accy[n] = zero; }

  for (int ks = 0; ks < EDIM / 32; ++ks) {
    v16bf a1 = ldA_f32(X1 + (size_t)m0 * EDIM + ks * 32, EDIM, lane);
    v16bf a2 = ldA_f32(X2 + (size_t)m0 * EDIM + ks * 32, EDIM, lane);
#pragma unroll
    for (int n = 0; n < 4; ++n) {
      v16bf bw1 = ldB_f32(W1 + (size_t)(h * 64 + n * 16) * EDIM + ks * 32, EDIM, lane);
      accx[n] = wmma_bf(a1, bw1, accx[n]);
      v16bf bw2 = ldB_f32(W2 + (size_t)(h * 64 + n * 16) * EDIM + ks * 32, EDIM, lane);
      accy[n] = wmma_bf(a2, bw2, accy[n]);
    }
  }

  int hhalf = (lane >> 4) & 1;
#pragma unroll
  for (int n = 0; n < 4; ++n) {
    int d = n * 16 + (lane & 15);
    float bx = b1[h * 64 + d], by = b2[h * 64 + d];
#pragma unroll
    for (int r = 0; r < 8; ++r) {
      int m = m0 + r + 8 * hhalf;
      int bb = m >> 11;            // S = 2048
      int srow = m & (SDIM - 1);
      float x = accx[n][r] + bx;
      float y = accy[n][r] + by;
      float rad = sqrtf(x * x + y * y);
      float c, s;
      if (rad > 0.f) { float inv = 1.f / rad; c = x * inv; s = y * inv; }
      else           { c = 1.f; s = 0.f; }   // atan2(0,0)=0 -> cos=1,sin=0
      bf16* row = feat + ((size_t)(bb * HDIM + h) * SDIM + srow) * FDIM + d;
      row[0]        = (__bf16)(cs * c);
      row[DDIM]     = (__bf16)(cs * s);
      row[2 * DDIM] = (__bf16)(rs * rad);
    }
  }
}

// ---------------- Kernel 2: V projection, stored transposed [B,H,D,S] ----------------
__global__ __launch_bounds__(128) void k_proj_vT(
    const float* __restrict__ X, const float* __restrict__ W,
    const float* __restrict__ bias, bf16* __restrict__ vt) {
  int lane = threadIdx.x & 31, wave = threadIdx.x >> 5;
  int m0 = blockIdx.x * 64 + wave * 16;
  int h  = blockIdx.y;
  v8f zero = {};
  v8f acc[4];
#pragma unroll
  for (int n = 0; n < 4; ++n) acc[n] = zero;

  for (int ks = 0; ks < EDIM / 32; ++ks) {
    v16bf a = ldA_f32(X + (size_t)m0 * EDIM + ks * 32, EDIM, lane);
#pragma unroll
    for (int n = 0; n < 4; ++n) {
      v16bf bw = ldB_f32(W + (size_t)(h * 64 + n * 16) * EDIM + ks * 32, EDIM, lane);
      acc[n] = wmma_bf(a, bw, acc[n]);
    }
  }

  int hhalf = (lane >> 4) & 1;
#pragma unroll
  for (int n = 0; n < 4; ++n) {
    int d = n * 16 + (lane & 15);
    float bv = bias[h * 64 + d];
#pragma unroll
    for (int r = 0; r < 8; ++r) {
      int m = m0 + r + 8 * hhalf;
      int bb = m >> 11;
      int srow = m & (SDIM - 1);
      vt[((size_t)(bb * HDIM + h) * DDIM + d) * SDIM + srow] = (__bf16)(acc[n][r] + bv);
    }
  }
}

// ---------------- Kernel 3: flash attention over polar features ----------------
// Qf/Kf: [B*H, S, FDIM] bf16 (scales folded into Qf). Vtx/Vty: [B*H, D, S] bf16.
// AX/AY: merged-head attention outputs [B, S, E] bf16.
// K/Vx/Vy tiles staged block-wide in LDS by the Tensor Data Mover (one wave each).
__global__ __launch_bounds__(128) void k_attn(
    const bf16* __restrict__ Qf, const bf16* __restrict__ Kf,
    const bf16* __restrict__ Vtx, const bf16* __restrict__ Vty,
    bf16* __restrict__ AX, bf16* __restrict__ AY) {
  __shared__ __align__(16) bf16 ktile[64 * FDIM];   // 24 KB: 64 keys x 192 feats
  __shared__ __align__(16) bf16 vxtile[64 * 64];    // 8 KB:  64 d    x 64 keys
  __shared__ __align__(16) bf16 vytile[64 * 64];    // 8 KB
  __shared__ __align__(16) bf16 pbuf[4][16 * 72];   // per-wave P staging, padded rows

  int lane = threadIdx.x & 31, wave = threadIdx.x >> 5;
  int bh = blockIdx.y;
  int q0 = blockIdx.x * 64 + wave * 16;
  const bf16* qbase = Qf + ((size_t)bh * SDIM + q0) * FDIM;
  const bf16* kbase = Kf + (size_t)bh * SDIM * FDIM;
  const bf16* vxh = Vtx + (size_t)bh * DDIM * SDIM;
  const bf16* vyh = Vty + (size_t)bh * DDIM * SDIM;
  bf16* plds = &pbuf[wave][0];
  int hhalf = (lane >> 4) & 1;

  v16bf qa[6];
#pragma unroll
  for (int ks = 0; ks < 6; ++ks) qa[ks] = ldA_bf16(qbase + ks * 32, FDIM, lane);

  v8f zero = {};
  v8f ox[4], oy[4];
#pragma unroll
  for (int t = 0; t < 4; ++t) { ox[t] = zero; oy[t] = zero; }
  float mrow[8], lrow[8];
#pragma unroll
  for (int r = 0; r < 8; ++r) { mrow[r] = -1e30f; lrow[r] = 0.f; }

  for (int kt = 0; kt < SDIM; kt += 64) {
    const bf16 *kS, *vxS, *vyS;
    int kLd, vLd;
#if USE_TDM
    // Stage shared tiles via TDM: K by wave0, Vx by wave1, Vy by wave2.
    if (wave == 0) {
      // rows = keys, row length 192 bf16 = 48 x 8B units, contiguous
      tdm_load_2d((unsigned)(uintptr_t)&ktile[0], kbase + (size_t)kt * FDIM,
                  /*t_d0=*/FDIM * 2 / 8, /*t_d1=*/64,
                  /*tile_d0=*/FDIM * 2 / 8, /*tile_d1=*/64,
                  /*stride_d0=*/FDIM * 2 / 8, /*log_ds=*/3);
    } else if (wave == 1) {
      // rows = d (64), row length 64 bf16 = 16 x 8B units, row stride S*2/8 = 512 units
      tdm_load_2d((unsigned)(uintptr_t)&vxtile[0], vxh + kt,
                  /*t_d0=*/16, /*t_d1=*/64, /*tile_d0=*/16, /*tile_d1=*/64,
                  /*stride_d0=*/SDIM * 2 / 8, /*log_ds=*/3);
    } else if (wave == 2) {
      tdm_load_2d((unsigned)(uintptr_t)&vytile[0], vyh + kt,
                  16, 64, 16, 64, SDIM * 2 / 8, 3);
    }
    __builtin_amdgcn_s_wait_tensorcnt(0);
    __syncthreads();                       // tiles visible block-wide
    kS = &ktile[0];  kLd = FDIM;
    vxS = &vxtile[0]; vyS = &vytile[0]; vLd = 64;
#else
    kS = kbase + (size_t)kt * FDIM; kLd = FDIM;
    vxS = vxh + kt; vyS = vyh + kt; vLd = SDIM;
#endif

    // ---- scores: 4 subtiles of 16 keys, K-dim 192 in 6 wmma steps ----
    v8f sc[4];
#pragma unroll
    for (int n = 0; n < 4; ++n) {
      v8f a = zero;
#pragma unroll
      for (int ks = 0; ks < 6; ++ks) {
        v16bf bk = ldB_bf16(kS + (size_t)(n * 16) * kLd + ks * 32, kLd, lane);
        a = wmma_bf(qa[ks], bk, a);
      }
      sc[n] = a;
    }
    // ---- online softmax (rows striped: row = r + 8*hhalf, cols across 16 lanes) ----
#pragma unroll
    for (int r = 0; r < 8; ++r) {
      float tm = fmaxf(fmaxf(sc[0][r], sc[1][r]), fmaxf(sc[2][r], sc[3][r]));
#pragma unroll
      for (int msk = 1; msk < 16; msk <<= 1) tm = fmaxf(tm, __shfl_xor(tm, msk, 32));
      float mnew = fmaxf(mrow[r], tm);
      float corr = __expf(mrow[r] - mnew);
      float ps = 0.f;
#pragma unroll
      for (int n = 0; n < 4; ++n) {
        float p = __expf(sc[n][r] - mnew);
        sc[n][r] = p;
        ps += p;
      }
#pragma unroll
      for (int msk = 1; msk < 16; msk <<= 1) ps += __shfl_xor(ps, msk, 32);
      lrow[r] = lrow[r] * corr + ps;
      mrow[r] = mnew;
#pragma unroll
      for (int t = 0; t < 4; ++t) { ox[t][r] *= corr; oy[t][r] *= corr; }
    }
    // ---- P: C-layout -> LDS -> A-fragment layout ----
#pragma unroll
    for (int n = 0; n < 4; ++n) {
      int col = n * 16 + (lane & 15);
#pragma unroll
      for (int r = 0; r < 8; ++r) plds[(r + 8 * hhalf) * 72 + col] = (__bf16)sc[n][r];
    }
    __builtin_amdgcn_wave_barrier();
    asm volatile("s_wait_dscnt 0" ::: "memory");   // same-wave LDS RAW fence
    v16bf pa0 = ldA_bf16(plds, 72, lane);          // keys kt+0..31
    v16bf pa1 = ldA_bf16(plds + 32, 72, lane);     // keys kt+32..63
    // ---- PV accumulation, vx and vy ----
#pragma unroll
    for (int t = 0; t < 4; ++t) {
      const bf16* vxb = vxS + (size_t)(t * 16) * vLd;
      const bf16* vyb = vyS + (size_t)(t * 16) * vLd;
      ox[t] = wmma_bf(pa0, ldB_bf16(vxb, vLd, lane), ox[t]);
      ox[t] = wmma_bf(pa1, ldB_bf16(vxb + 32, vLd, lane), ox[t]);
      oy[t] = wmma_bf(pa0, ldB_bf16(vyb, vLd, lane), oy[t]);
      oy[t] = wmma_bf(pa1, ldB_bf16(vyb + 32, vLd, lane), oy[t]);
    }
#if USE_TDM
    __syncthreads();                       // all reads done before next TDM overwrite
#else
    __builtin_amdgcn_wave_barrier();
#endif
  }

  // ---- normalize and write merged-head outputs ----
#pragma unroll
  for (int r = 0; r < 8; ++r) {
    float inv = 1.f / lrow[r];
#pragma unroll
    for (int t = 0; t < 4; ++t) { ox[t][r] *= inv; oy[t][r] *= inv; }
  }
  int b = bh / HDIM, h = bh % HDIM;
#pragma unroll
  for (int t = 0; t < 4; ++t) {
    int d = t * 16 + (lane & 15);
#pragma unroll
    for (int r = 0; r < 8; ++r) {
      int q = q0 + r + 8 * hhalf;
      size_t off = ((size_t)b * SDIM + q) * EDIM + h * DDIM + d;
      AX[off] = (__bf16)ox[t][r];
      AY[off] = (__bf16)oy[t][r];
    }
  }
}

// ---------------- Kernel 4: output projection (bf16 A, f32 W/out) ----------------
__global__ __launch_bounds__(128) void k_proj_out(
    const bf16* __restrict__ A, const float* __restrict__ W,
    const float* __restrict__ bias, float* __restrict__ out) {
  int lane = threadIdx.x & 31, wave = threadIdx.x >> 5;
  int m0 = blockIdx.x * 64 + wave * 16;
  int n0 = blockIdx.y * 64;
  v8f zero = {};
  v8f acc[4];
#pragma unroll
  for (int n = 0; n < 4; ++n) acc[n] = zero;

  for (int ks = 0; ks < EDIM / 32; ++ks) {
    v16bf a = ldA_bf16(A + (size_t)m0 * EDIM + ks * 32, EDIM, lane);
#pragma unroll
    for (int n = 0; n < 4; ++n) {
      v16bf bw = ldB_f32(W + (size_t)(n0 + n * 16) * EDIM + ks * 32, EDIM, lane);
      acc[n] = wmma_bf(a, bw, acc[n]);
    }
  }

  int hhalf = (lane >> 4) & 1;
#pragma unroll
  for (int n = 0; n < 4; ++n) {
    int col = n0 + n * 16 + (lane & 15);
    float bv = bias[col];
#pragma unroll
    for (int r = 0; r < 8; ++r) {
      int m = m0 + r + 8 * hhalf;
      out[(size_t)m * EDIM + col] = acc[n][r] + bv;
    }
  }
}

// ---------------- launch ----------------
extern "C" void kernel_launch(void* const* d_in, const int* in_sizes, int n_in,
                              void* d_out, int out_size, void* d_ws, size_t ws_size,
                              hipStream_t stream) {
  (void)in_sizes; (void)n_in; (void)out_size; (void)ws_size;
  const float* qx  = (const float*)d_in[0];
  const float* qy  = (const float*)d_in[1];
  const float* kx  = (const float*)d_in[2];
  const float* ky  = (const float*)d_in[3];
  const float* vx  = (const float*)d_in[4];
  const float* vy  = (const float*)d_in[5];
  const float* Wqx = (const float*)d_in[6];   const float* bqx = (const float*)d_in[7];
  const float* Wqy = (const float*)d_in[8];   const float* bqy = (const float*)d_in[9];
  const float* Wkx = (const float*)d_in[10];  const float* bkx = (const float*)d_in[11];
  const float* Wky = (const float*)d_in[12];  const float* bky = (const float*)d_in[13];
  const float* Wvx = (const float*)d_in[14];  const float* bvx = (const float*)d_in[15];
  const float* Wvy = (const float*)d_in[16];  const float* bvy = (const float*)d_in[17];
  const float* Wox = (const float*)d_in[18];  const float* box = (const float*)d_in[19];
  const float* Woy = (const float*)d_in[20];  const float* boy = (const float*)d_in[21];

  char* ws = (char*)d_ws;
  const size_t FEAT_BYTES = (size_t)BDIM * HDIM * SDIM * FDIM * sizeof(bf16); // 12.6 MB
  const size_t VT_BYTES   = (size_t)BDIM * HDIM * DDIM * SDIM * sizeof(bf16); // 4.2 MB
  const size_t AO_BYTES   = (size_t)BDIM * SDIM * EDIM * sizeof(bf16);        // 4.2 MB
  bf16* QF  = (bf16*)(ws);
  bf16* KF  = (bf16*)(ws + FEAT_BYTES);
  bf16* VTX = (bf16*)(ws + 2 * FEAT_BYTES);
  bf16* VTY = (bf16*)(ws + 2 * FEAT_BYTES + VT_BYTES);
  bf16* AX  = (bf16*)(ws + 2 * FEAT_BYTES + 2 * VT_BYTES);
  bf16* AY  = (bf16*)(ws + 2 * FEAT_BYTES + 2 * VT_BYTES + AO_BYTES);

  dim3 blk(128);
  dim3 gProj(BDIM * SDIM / 64, EDIM / 64);  // 64 x 8
  dim3 gAttn(SDIM / 64, BDIM * HDIM);       // 32 x 16

  const float cs = 0.5f / (float)DDIM;      // BETA / D
  const float rs = 0.5f / 8.0f;             // (1-BETA) / sqrt(D)

  k_proj_polar<<<gProj, blk, 0, stream>>>(qx, qy, Wqx, bqx, Wqy, bqy, QF, cs, rs);
  k_proj_polar<<<gProj, blk, 0, stream>>>(kx, ky, Wkx, bkx, Wky, bky, KF, 1.f, 1.f);
  k_proj_vT  <<<gProj, blk, 0, stream>>>(vx, Wvx, bvx, VTX);
  k_proj_vT  <<<gProj, blk, 0, stream>>>(vy, Wvy, bvy, VTY);
  k_attn     <<<gAttn, blk, 0, stream>>>(QF, KF, VTX, VTY, AX, AY);
  k_proj_out <<<gProj, blk, 0, stream>>>(AX, Wox, box, (float*)d_out);
  k_proj_out <<<gProj, blk, 0, stream>>>(AY, Woy, boy,
                                         (float*)d_out + (size_t)BDIM * SDIM * EDIM);
}
